// BuDingAttention_15144054686080
// MI455X (gfx1250) — matrix operand order repaired and checked by reading, not compiled, not run on hardware
//
#include <hip/hip_runtime.h>
#include <hip/hip_bf16.h>
#include <math.h>

// ---------------------------------------------------------------------------
// MI455X (gfx1250) GQA attention, WMMA f16 (f32 accumulate) end-to-end.
// ---------------------------------------------------------------------------

typedef __attribute__((ext_vector_type(16))) _Float16 v16h;
typedef __attribute__((ext_vector_type(8)))  _Float16 v8h;
typedef __attribute__((ext_vector_type(8)))  float    v8f;

static constexpr int  Bc   = 2;
static constexpr int  Sc   = 2048;
static constexpr int  Hc   = 2048;
static constexpr int  NHc  = 16;
static constexpr int  NKVc = 4;
static constexpr int  HDc  = 128;
static constexpr int  Gc   = NHc / NKVc;      // 4
static constexpr int  Mc   = Bc * Sc;         // 4096 rows of flattened tokens
static constexpr float SCALEc = 0.08838834764831845f;   // 1/sqrt(128)
static constexpr float LN1E4  = 9.210340371976184f;     // ln(10000)

// ---------------- WMMA fragment helpers (wave32, 16x16x32 f16) -------------
// A fragment (16x32, M x K), row-major source with row stride `ld` elements.
// Lane l: row m = l&15, half = l>>4.
//   slots 0..15 -> K = 16*half + i       (contiguous 32B)
__device__ __forceinline__ v16h load_a_tile(const _Float16* base, int ld, int lane) {
    const int m = lane & 15, half = lane >> 4;
    const _Float16* p = base + (size_t)m * ld + 16 * half;
    return *(const v16h*)(p);
}

// B fragment (32x16, K x N) from an N-major ("transposed") source:
// element (k, n) lives at baseT[n*ldk + k]. Lane l: n = l&15, K = 16*(l>>4)+slot
// -> one contiguous 32-byte load per lane.
__device__ __forceinline__ v16h load_bt_tile(const _Float16* baseT, int ldk, int lane) {
    const int n = lane & 15, half = lane >> 4;
    return *(const v16h*)(baseT + (size_t)n * ldk + 16 * half);
}

__device__ __forceinline__ v8f wmma_f16(v16h a, v16h b, v8f c) {
    return __builtin_amdgcn_wmma_f32_16x16x32_f16(
        /*neg_a=*/false, a, /*neg_b=*/false, b,
        /*c_mod=*/(short)0, c, /*reuse_a=*/false, /*reuse_b=*/false);
}

// ---------------- DPP16 16-lane butterfly reductions (no LDS) --------------
// DPP ctrl: 0xB1 = quad_perm [1,0,3,2] (xor 1), 0x4E = quad_perm [2,3,0,1]
// (xor 2), 0x141 = row_half_mirror (xor 7), 0x140 = row_mirror (xor 15).
// After xor1+xor2 each quad is uniform, so ^7 reduces bit 2 and ^15 bit 3.
// DPP16 rows are 16 lanes -> lanes 0-15 / 16-31 reduce independently,
// exactly matching the two C-layout halves.
template <int CTRL>
__device__ __forceinline__ float dpp_swap(float x) {
    return __builtin_bit_cast(float,
        __builtin_amdgcn_update_dpp(0, __builtin_bit_cast(int, x),
                                    CTRL, 0xf, 0xf, true));
}
__device__ __forceinline__ float rowmax16(float v) {
    v = fmaxf(v, dpp_swap<0xB1>(v));
    v = fmaxf(v, dpp_swap<0x4E>(v));
    v = fmaxf(v, dpp_swap<0x141>(v));
    v = fmaxf(v, dpp_swap<0x140>(v));
    return v;
}
__device__ __forceinline__ float rowsum16(float v) {
    v += dpp_swap<0xB1>(v);
    v += dpp_swap<0x4E>(v);
    v += dpp_swap<0x141>(v);
    v += dpp_swap<0x140>(v);
    return v;
}

// ---------------- prep kernels ---------------------------------------------
__global__ void k_cvt_f32_to_f16(const float* __restrict__ x,
                                 _Float16* __restrict__ y, int n) {
    int i = blockIdx.x * blockDim.x + threadIdx.x;
    if (i < n) y[i] = (_Float16)x[i];
}

// W (K x N, row-major f32) -> WT (N x K, row-major f16)
__global__ void k_transpose_to_f16(const float* __restrict__ W,
                                   _Float16* __restrict__ WT, int K, int N) {
    int i = blockIdx.x * blockDim.x + threadIdx.x;
    if (i >= K * N) return;
    int k = i / N, n = i - k * N;
    WT[(size_t)n * K + k] = (_Float16)W[i];
}

// ---------------- WMMA GEMM: C[M,N] = A[M,K] @ W[K,N] (+bias) ---------------
// A: f16 row-major; BT: f16 N-major (N x K). blockDim = 128 (4 waves),
// each wave owns a 32(M) x 64(N) tile (2 A-frags x 4 B-frags -> 8 WMMAs per
// 32-wide K chunk); grid = (N/64, M/128).
template <bool HASBIAS, bool F32OUT>
__global__ __launch_bounds__(128, 1)
void k_gemm_wmma(const _Float16* __restrict__ A,
                 const _Float16* __restrict__ BT,
                 const float* __restrict__ bias,
                 void* __restrict__ Cout,
                 int M, int N, int K) {
    const int lane = threadIdx.x & 31;
    const int wave = threadIdx.x >> 5;
    const int m0 = blockIdx.y * 128 + wave * 32;
    const int n0 = blockIdx.x * 64;

    v8f acc[2][4] = {};
    const _Float16* arow0 = A + (size_t)m0 * K;
    const _Float16* arow1 = arow0 + (size_t)16 * K;
    for (int kk = 0; kk < K; kk += 32) {
        __builtin_prefetch(arow0 + kk + 128, 0, 1);   // global_prefetch_b8
        const v16h a0 = load_a_tile(arow0 + kk, K, lane);
        const v16h a1 = load_a_tile(arow1 + kk, K, lane);
#pragma unroll
        for (int j = 0; j < 4; ++j) {
            const v16h b = load_bt_tile(BT + (size_t)(n0 + 16 * j) * K + kk, K, lane);
            acc[0][j] = wmma_f16(a0, b, acc[0][j]);
            acc[1][j] = wmma_f16(a1, b, acc[1][j]);
        }
    }

    const int n  = lane & 15;
    const int mb = (lane >> 4) * 8;   // C layout: vgpr r holds row mb + r
#pragma unroll
    for (int half = 0; half < 2; ++half) {
#pragma unroll
        for (int j = 0; j < 4; ++j) {
#pragma unroll
            for (int r = 0; r < 8; ++r) {
                const int m = m0 + 16 * half + mb + r;
                const int c = n0 + 16 * j + n;
                float v = acc[half][j][r];
                if (HASBIAS) v += bias[c];
                if (F32OUT) ((float*)Cout)[(size_t)m * N + c] = v;
                else        ((_Float16*)Cout)[(size_t)m * N + c] = (_Float16)v;
            }
        }
    }
}

// ---------------- RoPE + layout change -------------------------------------
// In : Ylin (B*S, nheads*HD) f16 (post-GEMM, bias included)
// Out: Yh   (B, nheads, S, HD) f16 with q' = q*cos + rotate_half(q)*sin
__global__ void k_rope(const _Float16* __restrict__ Ylin,
                       const int* __restrict__ pos_ids,
                       _Float16* __restrict__ Yh, int nheads) {
    int i = blockIdx.x * blockDim.x + threadIdx.x;     // over B*S*nheads*64
    const int total = Bc * Sc * nheads * (HDc / 2);
    if (i >= total) return;
    const int j = i & 63;            i >>= 6;          // rotary pair index
    const int h = i % nheads;        i /= nheads;
    const int s = i % Sc;
    const int b = i / Sc;
    const float pos = (float)pos_ids[b * Sc + s];
    const float inv_freq = __expf(-(float)j * (LN1E4 / 64.0f)); // 10000^(-2j/128)
    const float ang = pos * inv_freq;
    const float c = __cosf(ang), sn = __sinf(ang);
    const size_t src = ((size_t)(b * Sc + s) * nheads + h) * HDc;
    const float x1 = (float)Ylin[src + j];
    const float x2 = (float)Ylin[src + j + 64];
    const size_t dst = (((size_t)b * nheads + h) * Sc + s) * HDc;
    Yh[dst + j]      = (_Float16)(x1 * c - x2 * sn);
    Yh[dst + j + 64] = (_Float16)(x2 * c + x1 * sn);
}

// V (B*S, NKV*HD) f16 -> Vt (B, NKV, HD, S) f16  (so PV B-frags are contiguous)
__global__ void k_v_transpose(const _Float16* __restrict__ Vlin,
                              _Float16* __restrict__ Vt) {
    int i = blockIdx.x * blockDim.x + threadIdx.x;     // over B*S*NKV*HD
    if (i >= Bc * Sc * NKVc * HDc) return;
    const int d = i & (HDc - 1);     i >>= 7;
    const int h = i % NKVc;          i /= NKVc;
    const int s = i % Sc;
    const int b = i / Sc;
    Vt[(((size_t)b * NKVc + h) * HDc + d) * Sc + s] =
        Vlin[((size_t)(b * Sc + s) * NKVc + h) * HDc + d];
}

// ---------------- Flash attention (causal, online softmax) -----------------
// Q: (B,NH,S,HD), K: (B,NKV,S,HD), Vt: (B,NKV,HD,S), all f16.
// blockDim = 128 (4 independent waves); wave owns one 16-row Q tile.
// grid = (S/64, B*NH). Causal mask computed analytically; fully-masked K
// blocks are skipped (t0 < q0+16).
__global__ __launch_bounds__(128, 1)
void k_flash_attn(const _Float16* __restrict__ Qh,
                  const _Float16* __restrict__ Kh,
                  const _Float16* __restrict__ Vt,
                  _Float16* __restrict__ Ob) {
    __shared__ __attribute__((aligned(64))) _Float16 ldsP[4][16 * 32];

    const int lane = threadIdx.x & 31;
    const int wave = threadIdx.x >> 5;
    const int bh = blockIdx.y;
    const int b = bh / NHc, h = bh % NHc, kvh = h / Gc;
    const int q0 = blockIdx.x * 64 + wave * 16;

    const _Float16* Qp = Qh + ((size_t)(b * NHc  + h)   * Sc) * HDc;
    const _Float16* Kp = Kh + ((size_t)(b * NKVc + kvh) * Sc) * HDc;
    const _Float16* Vp = Vt + ((size_t)(b * NKVc + kvh) * HDc) * Sc;
    _Float16* lds = &ldsP[wave][0];

    // Q fragments: 4 K-chunks of 32 over HD=128
    v16h qf[4];
#pragma unroll
    for (int c = 0; c < 4; ++c)
        qf[c] = load_a_tile(Qp + (size_t)q0 * HDc + 32 * c, HDc, lane);

    v8f o[8] = {};                    // 16 x 128 f32 accumulator (C layout)
    float rm[8], rl[8];
#pragma unroll
    for (int r = 0; r < 8; ++r) { rm[r] = -3.0e38f; rl[r] = 0.0f; }

    const int n  = lane & 15;
    const int mb = (lane >> 4) * 8;

    for (int t0 = 0; t0 < q0 + 16; t0 += 32) {        // uniform per wave
        // ---- scores: two 16x16 tiles over K-chunks of HD ----
        v8f s0 = {}, s1 = {};
#pragma unroll
        for (int c = 0; c < 4; ++c) {
            const v16h kb0 = load_bt_tile(Kp + (size_t)t0 * HDc + 32 * c, HDc, lane);
            const v16h kb1 = load_bt_tile(Kp + (size_t)(t0 + 16) * HDc + 32 * c, HDc, lane);
            s0 = wmma_f16(qf[c], kb0, s0);
            s1 = wmma_f16(qf[c], kb1, s1);
        }
        // ---- scale + causal mask + online softmax (per C-layout row) ----
#pragma unroll
        for (int r = 0; r < 8; ++r) {
            const int m = q0 + mb + r;
            float v0 = s0[r] * SCALEc;
            float v1 = s1[r] * SCALEc;
            if (t0 + n > m)      v0 = -3.0e38f;
            if (t0 + 16 + n > m) v1 = -3.0e38f;
            const float mx   = rowmax16(fmaxf(v0, v1));
            const float mnew = fmaxf(rm[r], mx);
            const float corr = __expf(rm[r] - mnew);
            const float p0 = __expf(v0 - mnew);
            const float p1 = __expf(v1 - mnew);
            rl[r] = rl[r] * corr + rowsum16(p0 + p1);
            rm[r] = mnew;
#pragma unroll
            for (int j = 0; j < 8; ++j) o[j][r] *= corr;
            // stage P (16x32 row-major) in LDS for A-fragment reload
            lds[(mb + r) * 32 + n]      = (_Float16)p0;
            lds[(mb + r) * 32 + 16 + n] = (_Float16)p1;
        }
        // same-wave LDS RAW: DS ops are in-order per wave; fence the compiler
        // and drain DScnt explicitly.
        asm volatile("s_wait_dscnt 0x0" ::: "memory");

        // ---- O += P @ V : 8 WMMAs over the 128-wide head dim ----
        const v16h pf = load_a_tile(lds, 32, lane);
#pragma unroll
        for (int j = 0; j < 8; ++j) {
            const v16h vb = load_bt_tile(Vp + (size_t)(16 * j) * Sc + t0, Sc, lane);
            o[j] = wmma_f16(pf, vb, o[j]);
        }
    }

    // ---- normalize and scatter to Ob (B*S, NH*HD) f16 ----
#pragma unroll
    for (int j = 0; j < 8; ++j) {
#pragma unroll
        for (int r = 0; r < 8; ++r) {
            const int m = q0 + mb + r;
            const float val = o[j][r] / rl[r];
            Ob[((size_t)b * Sc + m) * (NHc * HDc) + h * HDc + 16 * j + n] =
                (_Float16)val;
        }
    }
}

// ---------------------------------------------------------------------------
extern "C" void kernel_launch(void* const* d_in, const int* in_sizes, int n_in,
                              void* d_out, int out_size, void* d_ws, size_t ws_size,
                              hipStream_t stream) {
    (void)in_sizes; (void)n_in; (void)out_size; (void)ws_size;

    const float* hidden = (const float*)d_in[0];
    const float* wq     = (const float*)d_in[1];
    const float* bq     = (const float*)d_in[2];
    const float* wk     = (const float*)d_in[3];
    const float* bk     = (const float*)d_in[4];
    const float* wv     = (const float*)d_in[5];
    const float* bv     = (const float*)d_in[6];
    const float* wo     = (const float*)d_in[7];
    /* d_in[8] = attention_mask: causal mask is computed analytically */
    const int*   pos    = (const int*)d_in[9];
    float* out = (float*)d_out;

    // ---- workspace carve-up (all chunks 256B aligned by construction) ----
    char* w = (char*)d_ws;
    size_t off = 0;
    auto alloc = [&](size_t bytes) -> void* {
        void* p = w + off; off += (bytes + 255) & ~(size_t)255; return p;
    };
    _Float16* Xh   = (_Float16*)alloc((size_t)Mc * Hc * 2);             // tokens f16
    _Float16* WqT  = (_Float16*)alloc((size_t)(NHc * HDc) * Hc * 2);    // 2048 x 2048
    _Float16* WkT  = (_Float16*)alloc((size_t)(NKVc * HDc) * Hc * 2);   // 512 x 2048
    _Float16* WvT  = (_Float16*)alloc((size_t)(NKVc * HDc) * Hc * 2);
    _Float16* WoT  = (_Float16*)alloc((size_t)Hc * (NHc * HDc) * 2);
    _Float16* Qlin = (_Float16*)alloc((size_t)Mc * NHc * HDc * 2);
    _Float16* Klin = (_Float16*)alloc((size_t)Mc * NKVc * HDc * 2);
    _Float16* Vlin = (_Float16*)alloc((size_t)Mc * NKVc * HDc * 2);
    _Float16* Qh   = (_Float16*)alloc((size_t)Mc * NHc * HDc * 2);
    _Float16* Kh   = (_Float16*)alloc((size_t)Mc * NKVc * HDc * 2);
    _Float16* Vt   = (_Float16*)alloc((size_t)Mc * NKVc * HDc * 2);
    _Float16* Ob   = Qlin;   // safe alias: Qlin dead after RoPE, Ob born after

    const int TPB = 256;
    auto blks = [](int n, int t) { return (n + t - 1) / t; };

    // 1) f32 -> f16 activations, transpose+convert weights to N-major
    { int n = Mc * Hc;
      k_cvt_f32_to_f16<<<blks(n, TPB), TPB, 0, stream>>>(hidden, Xh, n); }
    { int n = Hc * (NHc * HDc);
      k_transpose_to_f16<<<blks(n, TPB), TPB, 0, stream>>>(wq, WqT, Hc, NHc * HDc); }
    { int n = Hc * (NKVc * HDc);
      k_transpose_to_f16<<<blks(n, TPB), TPB, 0, stream>>>(wk, WkT, Hc, NKVc * HDc); }
    { int n = Hc * (NKVc * HDc);
      k_transpose_to_f16<<<blks(n, TPB), TPB, 0, stream>>>(wv, WvT, Hc, NKVc * HDc); }
    { int n = (NHc * HDc) * Hc;
      k_transpose_to_f16<<<blks(n, TPB), TPB, 0, stream>>>(wo, WoT, NHc * HDc, Hc); }

    // 2) QKV projections (WMMA, fused bias, f16 out)
    {
        dim3 g(NHc * HDc / 64, Mc / 128);
        k_gemm_wmma<true, false><<<g, 128, 0, stream>>>(Xh, WqT, bq, Qlin,
                                                        Mc, NHc * HDc, Hc);
    }
    {
        dim3 g(NKVc * HDc / 64, Mc / 128);
        k_gemm_wmma<true, false><<<g, 128, 0, stream>>>(Xh, WkT, bk, Klin,
                                                        Mc, NKVc * HDc, Hc);
        k_gemm_wmma<true, false><<<g, 128, 0, stream>>>(Xh, WvT, bv, Vlin,
                                                        Mc, NKVc * HDc, Hc);
    }

    // 3) RoPE (Q, K) + V transpose to (HD, S)
    { int n = Bc * Sc * NHc * (HDc / 2);
      k_rope<<<blks(n, TPB), TPB, 0, stream>>>(Qlin, pos, Qh, NHc); }
    { int n = Bc * Sc * NKVc * (HDc / 2);
      k_rope<<<blks(n, TPB), TPB, 0, stream>>>(Klin, pos, Kh, NKVc); }
    { int n = Bc * Sc * NKVc * HDc;
      k_v_transpose<<<blks(n, TPB), TPB, 0, stream>>>(Vlin, Vt); }

    // 4) causal flash attention (WMMA QK^T and PV, online softmax)
    {
        dim3 g(Sc / 64, Bc * NHc);
        k_flash_attn<<<g, 128, 0, stream>>>(Qh, Kh, Vt, Ob);
    }

    // 5) output projection, f32 result straight to d_out
    {
        dim3 g(Hc / 64, Mc / 128);
        k_gemm_wmma<false, true><<<g, 128, 0, stream>>>(Ob, WoT, nullptr, out,
                                                        Mc, Hc, NHc * HDc);
    }
}
